// BaseTransformerLayer_79551384256536
// MI455X (gfx1250) — matrix operand order repaired and checked by reading, not compiled, not run on hardware
//
#include <hip/hip_runtime.h>
#include <hip/hip_bf16.h>

// ---------------- problem constants (static per reference) ----------------
#define BS       8
#define NQ       900
#define NV       19560
#define EMBED    256
#define NHEAD    8
#define NLVL     4
#define NPTS     4
#define DH       32            // EMBED / NHEAD
#define MV       (BS * NV)     // 156480 rows of value
#define MQ       (BS * NQ)     // 7200 rows of query

// ---------------- WMMA vector types ----------------
typedef __attribute__((ext_vector_type(16))) __bf16 v16bf;
typedef __attribute__((ext_vector_type(8)))  float  v8f;

union FragBF { v16bf v; uint4 q[2]; };

__device__ inline unsigned short f2bf(float f) {
    unsigned u = __float_as_uint(f);
    u += 0x7fffu + ((u >> 16) & 1u);         // round-to-nearest-even
    return (unsigned short)(u >> 16);
}

// ---------------- convert f32 -> bf16 (flat) ----------------
__global__ void cvt_f32_bf16(const float* __restrict__ in,
                             unsigned short* __restrict__ out, long n) {
    long i = (long)blockIdx.x * blockDim.x + threadIdx.x;
    if (i < n) out[i] = f2bf(in[i]);
}

// ---------------- convert + transpose weights: W (KxN) -> Wt (NxK) bf16 ----
__global__ void cvt_transpose_bf16(const float* __restrict__ W,
                                   unsigned short* __restrict__ Wt,
                                   int K, int N) {
    int i = blockIdx.x * blockDim.x + threadIdx.x;
    if (i < K * N) {
        int n = i / K, k = i - n * K;
        Wt[i] = f2bf(W[(long)k * N + n]);
    }
}

// ---------------- wave-level WMMA GEMM ----------------
// C[M,N] = A[M,K](bf16,row-major) * B  with Bt[N,K](bf16, K contiguous)
//          + bias[N] (+ resid[M,N])
// one wave -> 16x64 output tile (four 16x16 f32 accumulators sharing A frag)
//
// Fragment layouts per CDNA5 ISA (wave32):
//   A 16x32: lane = M (mod 16); lanes 0-15 hold K {0..7,16..23},
//            lanes 16-31 hold K {8..15,24..31}  -> two 16B loads, +16 elem apart
//   B 32x16: lane = N (mod 16); lanes 0-15 hold K {0..7,16..23},
//            lanes 16-31 hold K {8..15,24..31}  -> two 16B loads, +16 elem apart
//   C/D 16x16: lane = N (mod 16); VGPR r -> M = (lane>>4)*8 + r
__global__ void gemm_bf16_wmma(const __bf16* __restrict__ A,
                               const __bf16* __restrict__ Bt,
                               const float*  __restrict__ bias,
                               const float*  __restrict__ resid,
                               float* __restrict__ C,
                               int M, int N, int K) {
    const int lane = threadIdx.x & 31;
    const int wave = (int)((blockIdx.x * blockDim.x + threadIdx.x) >> 5);
    const int tilesN = N >> 6;               // 64-wide tiles
    const int tm = wave / tilesN;
    const int tn = wave % tilesN;
    if (tm * 16 >= M) return;

    const int row = lane & 15;               // A: M-row / B: N-col within tile
    const int kbA = (lane >> 4) << 3;        // A K half base: 0 or 8
    const int kbB = (lane >> 4) << 3;        // B K half base: 0 or 8

    v8f acc0 = (v8f){0.f,0.f,0.f,0.f,0.f,0.f,0.f,0.f};
    v8f acc1 = acc0, acc2 = acc0, acc3 = acc0;

    const __bf16* arow = A + (long)(tm * 16 + row) * K + kbA;
    const __bf16* bc0  = Bt + (long)(tn * 64 +  0 + row) * K + kbB;
    const __bf16* bc1  = Bt + (long)(tn * 64 + 16 + row) * K + kbB;
    const __bf16* bc2  = Bt + (long)(tn * 64 + 32 + row) * K + kbB;
    const __bf16* bc3  = Bt + (long)(tn * 64 + 48 + row) * K + kbB;

    for (int k0 = 0; k0 < K; k0 += 32) {
        // hoist ALL fragment loads before any WMMA so the scheduler can keep
        // many b128 loads in flight and retire WMMAs on partial loadcnt waits
        FragBF a, b0, b1, b2, b3;
        a.q[0]  = *(const uint4*)(arow + k0);       // K = kbA+0..7
        a.q[1]  = *(const uint4*)(arow + k0 + 16);  // K = kbA+16..23
        b0.q[0] = *(const uint4*)(bc0 + k0);
        b0.q[1] = *(const uint4*)(bc0 + k0 + 16);
        b1.q[0] = *(const uint4*)(bc1 + k0);
        b1.q[1] = *(const uint4*)(bc1 + k0 + 16);
        b2.q[0] = *(const uint4*)(bc2 + k0);
        b2.q[1] = *(const uint4*)(bc2 + k0 + 16);
        b3.q[0] = *(const uint4*)(bc3 + k0);
        b3.q[1] = *(const uint4*)(bc3 + k0 + 16);
        __builtin_prefetch(arow + k0 + 64, 0, 1);   // global_prefetch_b8
        acc0 = __builtin_amdgcn_wmma_f32_16x16x32_bf16(
                   false, a.v, false, b0.v, (short)0, acc0, false, false);
        acc1 = __builtin_amdgcn_wmma_f32_16x16x32_bf16(
                   false, a.v, false, b1.v, (short)0, acc1, false, false);
        acc2 = __builtin_amdgcn_wmma_f32_16x16x32_bf16(
                   false, a.v, false, b2.v, (short)0, acc2, false, false);
        acc3 = __builtin_amdgcn_wmma_f32_16x16x32_bf16(
                   false, a.v, false, b3.v, (short)0, acc3, false, false);
    }

    const int mb = tm * 16 + ((lane >> 4) << 3);
    const int nn = tn * 64 + (lane & 15);
    v8f accs[4] = {acc0, acc1, acc2, acc3};
#pragma unroll
    for (int c = 0; c < 4; ++c) {
        const int n = nn + c * 16;
        const float bi = bias ? bias[n] : 0.f;
#pragma unroll
        for (int r = 0; r < 8; ++r) {
            long o = (long)(mb + r) * N + n;
            float v = accs[c][r] + bi;
            if (resid) v += resid[o];
            C[o] = v;
        }
    }
}

// ---------------- softmax + bilinear deformable sampling ----------------
// one wave per (b, q, h); lane = channel c in [0, 32)
// reads vproj f32 (B,NV,E), writes samp bf16 (B*Q, E)
__global__ void deform_sample(const float* __restrict__ vproj,
                              const float* __restrict__ refp,   // (B,Q,4,2)
                              const float* __restrict__ off,    // (B*Q,256)
                              const float* __restrict__ logit,  // (B*Q,128)
                              unsigned short* __restrict__ samp) {
    const int lane = threadIdx.x & 31;
    const int wave = (int)((blockIdx.x * blockDim.x + threadIdx.x) >> 5);
    if (wave >= MQ * NHEAD) return;
    const int h  = wave & (NHEAD - 1);
    const int bq = wave >> 3;
    const int b  = bq / NQ;

    // softmax over the 16 (l,p) logits for this (b,q,h) — lane-uniform
    const float* lg = logit + (long)bq * (NHEAD * NLVL * NPTS) + h * 16;
    float l[16], m = -3.402823466e38f;
#pragma unroll
    for (int i = 0; i < 16; ++i) { l[i] = lg[i]; m = fmaxf(m, l[i]); }
    float s = 0.f;
#pragma unroll
    for (int i = 0; i < 16; ++i) { l[i] = __expf(l[i] - m); s += l[i]; }
    const float inv = 1.f / s;

    const int   Hl_[4] = {92, 46, 23, 12};
    const int   Wl_[4] = {160, 80, 40, 20};
    const int   st_[4] = {0, 14720, 18400, 19320};

    const float* offp = off  + (long)bq * EMBED + h * 32;   // (l*8 + p*2 + d)
    const float* rp   = refp + (long)bq * NLVL * 2;
    const float* vb   = vproj + ((long)b * NV) * EMBED + h * DH + lane;

    float acc = 0.f;
#pragma unroll
    for (int lvl = 0; lvl < NLVL; ++lvl) {
        const float rx = rp[lvl * 2 + 0], ry = rp[lvl * 2 + 1];
        const int Hl = Hl_[lvl], Wl = Wl_[lvl];
        const float* vl = vb + (long)st_[lvl] * EMBED;
#pragma unroll
        for (int p = 0; p < NPTS; ++p) {
            const float ox = offp[lvl * 8 + p * 2 + 0];
            const float oy = offp[lvl * 8 + p * 2 + 1];
            // grid_sample align_corners=False pixel coords
            const float x = rx * Wl + ox - 0.5f;
            const float y = ry * Hl + oy - 0.5f;
            const float x0f = floorf(x), y0f = floorf(y);
            const int x0 = (int)x0f, y0 = (int)y0f;
            const float wx1 = x - x0f, wx0 = 1.f - wx1;
            const float wy1 = y - y0f, wy0 = 1.f - wy1;
            float sv = 0.f;
            // zero-padding: skip out-of-range taps (uniform branches per wave)
            if (x0 >= 0     && x0     < Wl && y0 >= 0     && y0     < Hl)
                sv += wx0 * wy0 * vl[(long)(y0 * Wl + x0) * EMBED];
            if (x0 + 1 >= 0 && x0 + 1 < Wl && y0 >= 0     && y0     < Hl)
                sv += wx1 * wy0 * vl[(long)(y0 * Wl + x0 + 1) * EMBED];
            if (x0 >= 0     && x0     < Wl && y0 + 1 >= 0 && y0 + 1 < Hl)
                sv += wx0 * wy1 * vl[(long)((y0 + 1) * Wl + x0) * EMBED];
            if (x0 + 1 >= 0 && x0 + 1 < Wl && y0 + 1 >= 0 && y0 + 1 < Hl)
                sv += wx1 * wy1 * vl[(long)((y0 + 1) * Wl + x0 + 1) * EMBED];
            acc += (l[lvl * 4 + p] * inv) * sv;
        }
    }
    samp[(long)bq * EMBED + h * DH + lane] = f2bf(acc);
}

// ---------------- launcher ----------------
extern "C" void kernel_launch(void* const* d_in, const int* in_sizes, int n_in,
                              void* d_out, int out_size, void* d_ws, size_t ws_size,
                              hipStream_t stream) {
    const float* query = (const float*)d_in[0];
    const float* value = (const float*)d_in[1];
    const float* refp  = (const float*)d_in[2];
    const float* Wv    = (const float*)d_in[3];
    const float* bv    = (const float*)d_in[4];
    const float* Ws    = (const float*)d_in[5];
    const float* bs    = (const float*)d_in[6];
    const float* Wa    = (const float*)d_in[7];
    const float* ba    = (const float*)d_in[8];
    const float* Wo    = (const float*)d_in[9];
    const float* bo    = (const float*)d_in[10];
    float* out = (float*)d_out;

    // ---- workspace partition (256B aligned) ----
    char* wsp = (char*)d_ws;
    auto take = [&](size_t bytes) -> void* {
        void* p = wsp; wsp += (bytes + 255) & ~(size_t)255; return p;
    };
    unsigned short* value_bf = (unsigned short*)take((size_t)MV * EMBED * 2);
    unsigned short* query_bf = (unsigned short*)take((size_t)MQ * EMBED * 2);
    unsigned short* samp_bf  = (unsigned short*)take((size_t)MQ * EMBED * 2);
    unsigned short* WvT = (unsigned short*)take((size_t)EMBED * EMBED * 2);
    unsigned short* WsT = (unsigned short*)take((size_t)EMBED * EMBED * 2);
    unsigned short* WaT = (unsigned short*)take((size_t)EMBED * 128 * 2);
    unsigned short* WoT = (unsigned short*)take((size_t)EMBED * EMBED * 2);
    float* vproj  = (float*)take((size_t)MV * EMBED * 4);
    float* offb   = (float*)take((size_t)MQ * EMBED * 4);
    float* logitb = (float*)take((size_t)MQ * 128 * 4);

    const int T = 256;
    auto gup = [](long n, int t) { return (unsigned)((n + t - 1) / t); };

    // 1) precision conversion
    cvt_f32_bf16<<<gup((long)MV * EMBED, T), T, 0, stream>>>(value, value_bf, (long)MV * EMBED);
    cvt_f32_bf16<<<gup((long)MQ * EMBED, T), T, 0, stream>>>(query, query_bf, (long)MQ * EMBED);
    cvt_transpose_bf16<<<gup(EMBED * EMBED, T), T, 0, stream>>>(Wv, WvT, EMBED, EMBED);
    cvt_transpose_bf16<<<gup(EMBED * EMBED, T), T, 0, stream>>>(Ws, WsT, EMBED, EMBED);
    cvt_transpose_bf16<<<gup(EMBED * 128,  T), T, 0, stream>>>(Wa, WaT, EMBED, 128);
    cvt_transpose_bf16<<<gup(EMBED * EMBED, T), T, 0, stream>>>(Wo, WoT, EMBED, EMBED);

    // 2) value projection: vproj = value @ Wv + bv   (dominant GEMM)
    {
        long waves = (long)(MV / 16) * (EMBED / 64);
        gemm_bf16_wmma<<<gup(waves * 32, T), T, 0, stream>>>(
            (const __bf16*)value_bf, (const __bf16*)WvT, bv, nullptr, vproj,
            MV, EMBED, EMBED);
    }
    // 3) sampling offsets: offb = query @ Ws + bs
    {
        long waves = (long)(MQ / 16) * (EMBED / 64);
        gemm_bf16_wmma<<<gup(waves * 32, T), T, 0, stream>>>(
            (const __bf16*)query_bf, (const __bf16*)WsT, bs, nullptr, offb,
            MQ, EMBED, EMBED);
    }
    // 4) attention logits: logitb = query @ Wa + ba
    {
        long waves = (long)(MQ / 16) * (128 / 64);
        gemm_bf16_wmma<<<gup(waves * 32, T), T, 0, stream>>>(
            (const __bf16*)query_bf, (const __bf16*)WaT, ba, nullptr, logitb,
            MQ, 128, EMBED);
    }
    // 5) softmax + bilinear deformable sampling -> samp_bf (bf16)
    {
        long waves = (long)MQ * NHEAD;
        deform_sample<<<gup(waves * 32, T), T, 0, stream>>>(
            vproj, refp, offb, logitb, samp_bf);
    }
    // 6) output projection + residual: out = samp @ Wo + bo + query
    {
        long waves = (long)(MQ / 16) * (EMBED / 64);
        gemm_bf16_wmma<<<gup(waves * 32, T), T, 0, stream>>>(
            (const __bf16*)samp_bf, (const __bf16*)WoT, bo, query, out,
            MQ, EMBED, EMBED);
    }
}
